// GNN_29188597744143
// MI455X (gfx1250) — matrix operand order repaired and checked by reading, not compiled, not run on hardware
//
#include <hip/hip_runtime.h>
#include <hip/hip_bf16.h>

typedef __attribute__((ext_vector_type(2)))  float    v2f;
typedef __attribute__((ext_vector_type(8)))  float    v8f;
typedef __attribute__((ext_vector_type(16))) _Float16 v16h;
typedef __attribute__((ext_vector_type(4)))  unsigned v4u;
typedef __attribute__((ext_vector_type(4)))  int      v4i;
typedef __attribute__((ext_vector_type(8)))  int      v8i;

#define DD 64
#define NEG_SLOPE 0.2f
#define EPS_SM 1e-16f
#define EPS_LN 1e-5f

#if __has_builtin(__builtin_amdgcn_wmma_f32_16x16x4_f32)
#define USE_WMMA_F32 1
#else
#define USE_WMMA_F32 0
#endif

#if defined(__gfx1250__) && __has_builtin(__builtin_amdgcn_tensor_load_to_lds)
#define HAVE_TDM 1
#else
#define HAVE_TDM 0
#endif

// ---------- ordered-uint encoding of float for atomic max ----------
__device__ __forceinline__ unsigned fenc(float f) {
    unsigned u = __float_as_uint(f);
    return (u & 0x80000000u) ? ~u : (u | 0x80000000u);
}
__device__ __forceinline__ float fdec(unsigned u) {
    return __uint_as_float((u & 0x80000000u) ? (u & 0x7fffffffu) : ~u);
}

#if HAVE_TDM
// Issue a TDM DMA of one 64x64 f32 matrix (row-major, stride 64) into LDS.
// D# per CDNA5 ISA section 8: group0 = {count, lds_addr, global_addr, type=2},
// group1 = {data_size=4B, tensor_dim0/1=64, tile_dim0/1=64, dim0_stride=64,
//           dim1_stride=4096}, groups 2/3 zero (2D tensor).
__device__ __forceinline__ void tdm_load_w(const float* gW, unsigned ldsOff) {
    unsigned long long ga = (unsigned long long)(size_t)gW;
    v4u g0 = { 1u,                                   // count=1, is_restore=0
               ldsOff,                               // lds_addr (bytes)
               (unsigned)(ga & 0xffffffffu),         // global_addr[31:0]
               (unsigned)((ga >> 32) & 0x1ffffffu) | 0x80000000u }; // addr[56:32] | type=2
    v8i g1 = { 0x20000,        // data_size=2 (4 bytes)
               64 << 16,       // tensor_dim0[15:0]
               64 << 16,       // tensor_dim1[15:0] (dim0 hi16 = 0)
               64 << 16,       // tile_dim0
               64,             // tile_dim1 (tile_dim2 = 0)
               64,             // tensor_dim0_stride[31:0]
               0x10000000,     // tensor_dim1_stride[15:0] = 4096 (stride0 hi16 = 0)
               0 };
    v4i z4 = { 0, 0, 0, 0 };
#if defined(__clang_major__) && (__clang_major__ >= 23)
    v8i z8 = { 0, 0, 0, 0, 0, 0, 0, 0 };
    __builtin_amdgcn_tensor_load_to_lds(g0, g1, z4, z4, z8, 0);
#else
    __builtin_amdgcn_tensor_load_to_lds(g0, g1, z4, z4, 0);
#endif
}
__device__ __forceinline__ void tdm_wait0() {
#if __has_builtin(__builtin_amdgcn_s_wait_tensorcnt)
    __builtin_amdgcn_s_wait_tensorcnt(0);
#else
    asm volatile("s_wait_tensorcnt 0x0" ::: "memory");
#endif
}
__device__ __forceinline__ unsigned lds_offset_of(const float* p) {
    // generic -> LDS addrspace cast, then ptrtoint => raw LDS byte offset
    __attribute__((address_space(3))) const float* lp =
        (__attribute__((address_space(3))) const float*)p;
    return (unsigned)(size_t)lp;
}
#endif

// ---------- fused multi-weight GEMM: O_i = A @ W_i  (A:[N,64], W:[64,64]) ----------
// Block = 8 waves = 128 rows. Weights staged once per block into LDS via TDM
// (fallback: cooperative vector copy), B-fragments served from LDS.
__global__ __launch_bounds__(256) void gat_gemm_wmma(
    const float* __restrict__ A, int N,
    const float* __restrict__ W0, const float* __restrict__ W1, const float* __restrict__ W2,
    float* __restrict__ O0, float* __restrict__ O1, float* __restrict__ O2)
{
    __shared__ float sW[3 * 64 * 64];   // 48 KB of the 320 KB WGP LDS

    const float* Ws[3] = { W0, W1, W2 };
    float*       Os[3] = { O0, O1, O2 };

#if HAVE_TDM
    if ((threadIdx.x >> 5) == 0) {      // one wave issues the DMAs
#pragma unroll
        for (int w = 0; w < 3; ++w)
            if (Ws[w]) tdm_load_w(Ws[w], lds_offset_of(&sW[w * 4096]));
        tdm_wait0();
    }
#else
#pragma unroll
    for (int w = 0; w < 3; ++w) {
        if (Ws[w]) {
            const float4* s4 = (const float4*)Ws[w];
            float4*       d4 = (float4*)&sW[w * 4096];
            for (int i = threadIdx.x; i < 1024; i += 256) d4[i] = s4[i];
        }
    }
#endif
    __syncthreads();

    const int lane = threadIdx.x & 31;
    const int wv   = threadIdx.x >> 5;
    const int rowBase = (blockIdx.x * 8 + wv) * 16;
    if (rowBase >= N) return;
    const int m  = lane & 15;   // M index (and N index for B/C)
    const int hh = lane >> 4;   // lane half

    const float* __restrict__ arow = A + (size_t)(rowBase + m) * DD;

#if USE_WMMA_F32
    // A fragment: 16x4 f32 tiles, 16 K-steps; contiguous K-pair -> b64 load.
    v2f af[16];
    const float2* __restrict__ ar2 = (const float2*)arow;
#pragma unroll
    for (int kk = 0; kk < 16; ++kk) {
        float2 p = ar2[kk * 2 + hh];
        af[kk][0] = p.x;
        af[kk][1] = p.y;
    }
#else
    v16h af[2];
#pragma unroll
    for (int t = 0; t < 2; ++t)
#pragma unroll
        for (int j = 0; j < 16; ++j) {
            int K = t * 32 + j + 8 * hh + ((j >= 8) ? 8 : 0);
            af[t][j] = (_Float16)arow[K];
        }
#endif

#pragma unroll   // full unroll: specialize per weight, no runtime pointer selects
    for (int w = 0; w < 3; ++w) {
        if (Ws[w] == nullptr) continue;
        const float* sWw = &sW[w * 4096];
        float* __restrict__ O = Os[w];
#pragma unroll
        for (int c = 0; c < 4; ++c) {
            v8f acc = {0.f, 0.f, 0.f, 0.f, 0.f, 0.f, 0.f, 0.f};
#if USE_WMMA_F32
#pragma unroll
            for (int kk = 0; kk < 16; ++kk) {
                const float* wp = sWw + (kk * 4 + 2 * hh) * DD + c * 16 + m;
                v2f b; b[0] = wp[0]; b[1] = wp[DD];
                acc = __builtin_amdgcn_wmma_f32_16x16x4_f32(
                    false, af[kk], false, b, (short)0, acc, false, false);
            }
#else
#pragma unroll
            for (int t = 0; t < 2; ++t) {
                v16h b;
#pragma unroll
                for (int j = 0; j < 16; ++j)
                    b[j] = (_Float16)sWw[(t * 32 + j + 8 * hh + ((j >= 8) ? 8 : 0)) * DD + c * 16 + m];
                acc = __builtin_amdgcn_wmma_f32_16x16x32_f16(
                    false, af[t], false, b, (short)0, acc, false, false);
            }
#endif
            // C/D layout: VGPR v -> row (v + 8*half), col = lane&15
            float* __restrict__ orow = O + (size_t)(rowBase + 8 * hh) * DD + c * 16 + m;
#pragma unroll
            for (int v = 0; v < 8; ++v) orow[(size_t)v * DD] = acc[v];
        }
    }
}

// ---------- per-node attention dots: o1 = M1 . a1, o2 = M2 . a2 (wave per node) ----------
__global__ __launch_bounds__(256) void gat_dots(
    const float* __restrict__ M1, const float* __restrict__ a1, float* __restrict__ o1,
    const float* __restrict__ M2, const float* __restrict__ a2, float* __restrict__ o2,
    int N)
{
    const int lane = threadIdx.x & 31;
    const int n = blockIdx.x * 8 + (threadIdx.x >> 5);
    if (n >= N) return;
    const int d0 = lane * 2;
    float2 v1 = ((const float2*)(M1 + (size_t)n * DD))[lane];
    float2 v2 = ((const float2*)(M2 + (size_t)n * DD))[lane];
    float p1 = v1.x * a1[d0] + v1.y * a1[d0 + 1];
    float p2 = v2.x * a2[d0] + v2.y * a2[d0 + 1];
#pragma unroll
    for (int off = 16; off > 0; off >>= 1) {
        p1 += __shfl_xor(p1, off, 32);
        p2 += __shfl_xor(p2, off, 32);
    }
    if (lane == 0) { o1[n] = p1; o2[n] = p2; }
}

// ---------- per-layer state init (acc, denom, running max, LN scalars), float4 ----------
__global__ __launch_bounds__(256) void gat_init(
    float4* __restrict__ acc4, float* __restrict__ denom,
    unsigned* __restrict__ mEnc, float* __restrict__ scal, int N)
{
    const int t = blockIdx.x * blockDim.x + threadIdx.x;
    if (t < N * (DD / 4)) acc4[t] = make_float4(0.f, 0.f, 0.f, 0.f);
    if (t < N) { denom[t] = 0.f; mEnc[t] = 0u; }
    if (t < 2) scal[t] = 0.f;
}

// ---------- edge pass 1: logit + leakyrelu + segment max ----------
__global__ __launch_bounds__(256) void gat_edge1(
    const int* __restrict__ src, const int* __restrict__ dst,
    const float* __restrict__ aS, const float* __restrict__ aD,
    float* __restrict__ edgeA, unsigned* __restrict__ mEnc, int E)
{
    const int e = blockIdx.x * blockDim.x + threadIdx.x;
    if (e >= E) return;
    float al = aS[src[e]] + aD[dst[e]];
    al = (al >= 0.f) ? al : NEG_SLOPE * al;
    edgeA[e] = al;
    atomicMax(&mEnc[dst[e]], fenc(al));
}

// ---------- edge pass 2 (grid-stride wave-per-edge + next-gather prefetch):
//            e = exp(a - m[dst]); denom += e; acc[dst] += e * x_src[src] ----------
__global__ __launch_bounds__(256) void gat_edge2(
    const int* __restrict__ src, const int* __restrict__ dst,
    const float* __restrict__ edgeA, const unsigned* __restrict__ mEnc,
    const float* __restrict__ xs,
    float* __restrict__ acc, float* __restrict__ denom, int E)
{
    const int lane   = threadIdx.x & 31;
    const int wave   = blockIdx.x * 8 + (threadIdx.x >> 5);
    const int nWaves = gridDim.x * 8;
    for (int e = wave; e < E; e += nWaves) {
        const int s = src[e], d = dst[e];
        const int en = e + nWaves;
        if (en < E) {  // hide next gather behind this edge's atomics
            __builtin_prefetch(xs + (size_t)src[en] * DD + lane * 2, 0, 0);
        }
        const float ev = __expf(edgeA[e] - fdec(mEnc[d]));
        if (lane == 0) atomicAdd(&denom[d], ev);
        const float2 xv = ((const float2*)(xs + (size_t)s * DD))[lane];
        float* ap = acc + (size_t)d * DD + lane * 2;
        atomicAdd(ap + 0, ev * xv.x);
        atomicAdd(ap + 1, ev * xv.y);
    }
}

// ---------- node finalize (float4): h = acc/(denom+eps) + bias + lin + lin_b (, ReLU) ----------
__global__ __launch_bounds__(256) void gat_final(
    const float4* __restrict__ acc4, const float* __restrict__ denom,
    const float* __restrict__ bias, const float4* __restrict__ lin4,
    const float* __restrict__ lin_b, float4* __restrict__ h4, int N, int relu)
{
    const int t = blockIdx.x * blockDim.x + threadIdx.x;   // one float4 per thread
    if (t >= N * (DD / 4)) return;
    const int n = t >> 4;            // 16 float4 per node
    const int d = (t & 15) * 4;      // feature base
    const float inv = 1.f / (denom[n] + EPS_SM);
    float4 a = acc4[t], l = lin4[t];
    float4 v;
    v.x = a.x * inv + bias[d + 0] + l.x + lin_b[d + 0];
    v.y = a.y * inv + bias[d + 1] + l.y + lin_b[d + 1];
    v.z = a.z * inv + bias[d + 2] + l.z + lin_b[d + 2];
    v.w = a.w * inv + bias[d + 3] + l.w + lin_b[d + 3];
    if (relu) {
        v.x = fmaxf(v.x, 0.f); v.y = fmaxf(v.y, 0.f);
        v.z = fmaxf(v.z, 0.f); v.w = fmaxf(v.w, 0.f);
    }
    h4[t] = v;
}

// ---------- global sum / sumsq for graph LayerNorm ----------
__global__ __launch_bounds__(256) void gat_reduce(
    const float* __restrict__ h, int total, float* __restrict__ scal)
{
    float s = 0.f, q = 0.f;
    for (int i = blockIdx.x * blockDim.x + threadIdx.x; i < total;
         i += gridDim.x * blockDim.x) {
        float v = h[i]; s += v; q += v * v;
    }
#pragma unroll
    for (int off = 16; off > 0; off >>= 1) {
        s += __shfl_xor(s, off, 32);
        q += __shfl_xor(q, off, 32);
    }
    if ((threadIdx.x & 31) == 0) { atomicAdd(&scal[0], s); atomicAdd(&scal[1], q); }
}

// ---------- LayerNorm + final projection to [N,1] (wave per node) ----------
__global__ __launch_bounds__(256) void gat_proj(
    const float* __restrict__ h, const float* __restrict__ scal,
    const float* __restrict__ lnw, const float* __restrict__ lnb,
    const float* __restrict__ pW, const float* __restrict__ pb,
    float* __restrict__ out, int N)
{
    const int lane = threadIdx.x & 31;
    const int n = blockIdx.x * 8 + (threadIdx.x >> 5);
    if (n >= N) return;
    const float inv = 1.f / ((float)N * (float)DD);
    const float mu  = scal[0] * inv;
    const float var = scal[1] * inv - mu * mu;
    const float rstd = rsqrtf(var + EPS_LN);
    const int d0 = lane * 2;
    const float2 hv = ((const float2*)(h + (size_t)n * DD))[lane];
    float g = ((hv.x - mu) * rstd * lnw[d0]     + lnb[d0])     * pW[d0]
            + ((hv.y - mu) * rstd * lnw[d0 + 1] + lnb[d0 + 1]) * pW[d0 + 1];
#pragma unroll
    for (int off = 16; off > 0; off >>= 1) g += __shfl_xor(g, off, 32);
    if (lane == 0) out[n] = g + pb[0];
}

extern "C" void kernel_launch(void* const* d_in, const int* in_sizes, int n_in,
                              void* d_out, int out_size, void* d_ws, size_t ws_size,
                              hipStream_t stream) {
    (void)n_in; (void)out_size; (void)ws_size;
    const int N = in_sizes[0] / DD;
    const int E = in_sizes[1] / 2;

    const float* x      = (const float*)d_in[0];
    const int*   ei     = (const int*)d_in[1];
    const int*   src    = ei;
    const int*   dst    = ei + E;
    const float* W_src0 = (const float*)d_in[2];
    const float* W_dst0 = (const float*)d_in[3];
    const float* a_src0 = (const float*)d_in[4];
    const float* a_dst0 = (const float*)d_in[5];
    const float* b0     = (const float*)d_in[6];
    const float* lin_W0 = (const float*)d_in[7];
    const float* lin_b0 = (const float*)d_in[8];
    const float* W1     = (const float*)d_in[9];
    const float* a_src1 = (const float*)d_in[10];
    const float* a_dst1 = (const float*)d_in[11];
    const float* b1     = (const float*)d_in[12];
    const float* lin_W1 = (const float*)d_in[13];
    const float* lin_b1 = (const float*)d_in[14];
    const float* W2     = (const float*)d_in[15];
    const float* a_src2 = (const float*)d_in[16];
    const float* a_dst2 = (const float*)d_in[17];
    const float* b2     = (const float*)d_in[18];
    const float* lin_W2 = (const float*)d_in[19];
    const float* lin_b2 = (const float*)d_in[20];
    const float* ln_w   = (const float*)d_in[21];
    const float* ln_b   = (const float*)d_in[22];
    const float* proj_W = (const float*)d_in[23];
    const float* proj_b = (const float*)d_in[24];
    float* out = (float*)d_out;

    // ---- workspace carve (floats) ----
    float* F      = (float*)d_ws;
    float* hbuf   = F;                               // [N,64]
    float* tmpA   = hbuf   + (size_t)N * DD;         // [N,64] src projection
    float* tmpB   = tmpA   + (size_t)N * DD;         // [N,64] dst projection (layer 0)
    float* tmpLin = tmpB   + (size_t)N * DD;         // [N,64] skip GEMM
    float* acc    = tmpLin + (size_t)N * DD;         // [N,64] scatter accumulator
    float* alphaS = acc    + (size_t)N * DD;         // [N]
    float* alphaD = alphaS + N;                      // [N]
    float* denom  = alphaD + N;                      // [N]
    unsigned* mEnc = (unsigned*)(denom + N);         // [N]
    float* edgeA  = (float*)(mEnc + N);              // [E]
    float* scal   = edgeA + E;                       // [2] sum, sumsq

    const dim3 blk(256);
    const int gV4   = (N * (DD / 4) + 255) / 256;    // float4-granular passes
    const int gEdge  = (E + 255) / 256;
    const int gNodeW = (N + 7) / 8;
    const int gEdge2 = 2048;                 // grid-stride wave-per-edge
    const int gGemm  = (N + 127) / 128;      // 8 waves x 16 rows per block

    // ---------------- Layer 0 ----------------
    gat_gemm_wmma<<<gGemm, blk, 0, stream>>>(x, N, W_src0, W_dst0, lin_W0, tmpA, tmpB, tmpLin);
    gat_dots<<<gNodeW, blk, 0, stream>>>(tmpA, a_src0, alphaS, tmpB, a_dst0, alphaD, N);
    gat_init<<<gV4, blk, 0, stream>>>((float4*)acc, denom, mEnc, scal, N);
    gat_edge1<<<gEdge, blk, 0, stream>>>(src, dst, alphaS, alphaD, edgeA, mEnc, E);
    gat_edge2<<<gEdge2, blk, 0, stream>>>(src, dst, edgeA, mEnc, tmpA, acc, denom, E);
    gat_final<<<gV4, blk, 0, stream>>>((const float4*)acc, denom, b0,
                                       (const float4*)tmpLin, lin_b0, (float4*)hbuf, N, 1);

    // ---------------- Layer 1 ----------------
    gat_gemm_wmma<<<gGemm, blk, 0, stream>>>(hbuf, N, W1, lin_W1, nullptr, tmpA, tmpLin, nullptr);
    gat_dots<<<gNodeW, blk, 0, stream>>>(tmpA, a_src1, alphaS, tmpA, a_dst1, alphaD, N);
    gat_init<<<gV4, blk, 0, stream>>>((float4*)acc, denom, mEnc, scal, N);
    gat_edge1<<<gEdge, blk, 0, stream>>>(src, dst, alphaS, alphaD, edgeA, mEnc, E);
    gat_edge2<<<gEdge2, blk, 0, stream>>>(src, dst, edgeA, mEnc, tmpA, acc, denom, E);
    gat_final<<<gV4, blk, 0, stream>>>((const float4*)acc, denom, b1,
                                       (const float4*)tmpLin, lin_b1, (float4*)hbuf, N, 1);

    // ---------------- Layer 2 ----------------
    gat_gemm_wmma<<<gGemm, blk, 0, stream>>>(hbuf, N, W2, lin_W2, nullptr, tmpA, tmpLin, nullptr);
    gat_dots<<<gNodeW, blk, 0, stream>>>(tmpA, a_src2, alphaS, tmpA, a_dst2, alphaD, N);
    gat_init<<<gV4, blk, 0, stream>>>((float4*)acc, denom, mEnc, scal, N);
    gat_edge1<<<gEdge, blk, 0, stream>>>(src, dst, alphaS, alphaD, edgeA, mEnc, E);
    gat_edge2<<<gEdge2, blk, 0, stream>>>(src, dst, edgeA, mEnc, tmpA, acc, denom, E);
    gat_final<<<gV4, blk, 0, stream>>>((const float4*)acc, denom, b2,
                                       (const float4*)tmpLin, lin_b2, (float4*)hbuf, N, 0);

    // ---------------- LayerNorm (graph) + projection ----------------
    gat_reduce<<<1024, blk, 0, stream>>>(hbuf, N * DD, scal);
    gat_proj<<<gNodeW, blk, 0, stream>>>(hbuf, scal, ln_w, ln_b, proj_W, proj_b, out, N);
}
